// S2Conv_80779744903848
// MI455X (gfx1250) — matrix operand order repaired and checked by reading, not compiled, not run on hardware
//
#include <hip/hip_runtime.h>

// ---------------------------------------------------------------------------
// S2 spherical convolution for MI455X (gfx1250, wave32, WMMA).
//
// Pass 0: split x, w, Y into bf16 hi/lo planes (one-time, ~25 MB in d_ws).
// Stage 1: psi[(e*64+g), i] = (1/sqrt(512)) * sum_n w[e,g,n] * Y[n,i]
//          GEMM M=4096, N=121, K=512 -> psi written as bf16 hi/lo planes.
// Stage 2: per l (d=2l+1):
//          out[b,g, off + u*d + m] = (1/8) * sum_e psi[e,g,l^2+u]*x[b,e,l^2+m]
//          GEMM M=64d (g,u), N=512d (b,m), K=64; 32x32 macro-tile per wave.
//
// Precision: bf16 hi/lo split (hi*hi + hi*lo + lo*hi, f32 accumulate) via
// v_wmma_f32_16x16x32_bf16 -> ~fp32 accuracy. Hot loops contain only loads +
// WMMA. OOB GEMM columns are merely clamped (not zeroed): C columns are
// independent, and the owning lanes skip their stores, so no masking VALU
// and no load serialization.
// Output stream (232 MB fp32) bounds runtime: ~10 us at 23.3 TB/s.
// ---------------------------------------------------------------------------

#define F_IN    64
#define F_OUT   64
#define N_GRID  512
#define NSH     121     // (LMAX+1)^2
#define OUTC    1771    // sum_{l=0..10} (2l+1)^2

#define XN      (512 * F_IN * NSH)      // 3,964,928 elements
#define WN      (F_IN * F_OUT * N_GRID) // 2,097,152
#define YN      (N_GRID * NSH)          // 61,952
#define PSIN    (F_IN * F_OUT * NSH)    // 495,616

typedef __attribute__((ext_vector_type(16))) __bf16          v16bf;
typedef __attribute__((ext_vector_type(16))) unsigned short  v16u;
typedef __attribute__((ext_vector_type(8)))  unsigned short  v8u;
typedef __attribute__((ext_vector_type(8)))  float           v8f;

static __device__ __forceinline__ unsigned short bf16_rne(float f) {
  unsigned u = __builtin_bit_cast(unsigned, f);
  u += 0x7FFFu + ((u >> 16) & 1u);          // round-to-nearest-even
  return (unsigned short)(u >> 16);
}

static __device__ __forceinline__ void bf16_split(float f, unsigned short& hi,
                                                  unsigned short& lo) {
  hi = bf16_rne(f);
  float fh = __builtin_bit_cast(float, ((unsigned)hi) << 16);
  lo = bf16_rne(f - fh);
}

static __device__ __forceinline__ v8f wmma3(v8f c, const v16u& ahi, const v16u& alo,
                                            const v16u& bhi, const v16u& blo) {
  const v16bf Ah = __builtin_bit_cast(v16bf, ahi);
  const v16bf Al = __builtin_bit_cast(v16bf, alo);
  const v16bf Bh = __builtin_bit_cast(v16bf, bhi);
  const v16bf Bl = __builtin_bit_cast(v16bf, blo);
  c = __builtin_amdgcn_wmma_f32_16x16x32_bf16(false, Ah, false, Bh, (short)0, c, false, false);
  c = __builtin_amdgcn_wmma_f32_16x16x32_bf16(false, Ah, false, Bl, (short)0, c, false, false);
  c = __builtin_amdgcn_wmma_f32_16x16x32_bf16(false, Al, false, Bh, (short)0, c, false, false);
  return c;
}

// ---------------------------------------------------------------------------
// Pass 0: fp32 -> bf16 hi/lo planes.
// ---------------------------------------------------------------------------
__global__ __launch_bounds__(256) void split_kernel(const float* __restrict__ src,
                                                    unsigned short* __restrict__ hi,
                                                    unsigned short* __restrict__ lo,
                                                    int n) {
  const int i = blockIdx.x * 256 + threadIdx.x;
  if (i < n) {
    unsigned short h, l;
    bf16_split(src[i], h, l);
    hi[i] = h; lo[i] = l;
  }
}

// ---------------------------------------------------------------------------
// Stage 1: psi = (w as 4096x512) x (Y as 512x121) / sqrt(512), bf16-plane IO.
// 4 waves per block, 1 wave per 16x16 output tile. Grid: (64, 8).
// ---------------------------------------------------------------------------
__global__ __launch_bounds__(128) void psi_gemm_kernel(const unsigned short* __restrict__ whi,
                                                       const unsigned short* __restrict__ wlo,
                                                       const unsigned short* __restrict__ yhi,
                                                       const unsigned short* __restrict__ ylo,
                                                       unsigned short* __restrict__ phi,
                                                       unsigned short* __restrict__ plo) {
  const int lane = threadIdx.x & 31;
  const int wv   = threadIdx.x >> 5;
  const int rt   = blockIdx.x * 4 + wv;      // row tile 0..255
  const int ct   = blockIdx.y;               // col tile 0..7
  const int half = lane >> 4;
  const int lx   = lane & 15;

  const int arow  = rt * 16 + lx;            // always in range (4096 rows)
  const int bcol  = ct * 16 + lx;            // column i, may exceed 120
  const bool bok  = (bcol < NSH);
  const int bcolc = bok ? bcol : (NSH - 1);  // clamp: loads legal, garbage
                                             // columns are never stored.

  const unsigned short* wh = whi + (size_t)arow * N_GRID;
  const unsigned short* wl = wlo + (size_t)arow * N_GRID;
  const unsigned short* yh = yhi + bcolc;
  const unsigned short* yl = ylo + bcolc;

  v8f c = {};
#pragma unroll 2
  for (int kt = 0; kt < 16; ++kt) {          // K = 512 = 16 * 32
    const int kbase = kt * 32;
    // A fragment: element j -> K = kbase + (j&7) + ((j&8)<<1) + 8*half
    // => two contiguous 8-bf16 runs per lane: one b128 load each.
    const int r0 = kbase + (half << 3);
    const int r1 = kbase + 16 + (half << 3);
    const v8u ah0 = *reinterpret_cast<const v8u*>(wh + r0);
    const v8u ah1 = *reinterpret_cast<const v8u*>(wh + r1);
    const v8u al0 = *reinterpret_cast<const v8u*>(wl + r0);
    const v8u al1 = *reinterpret_cast<const v8u*>(wl + r1);

    v16u ahi, alo, bhi, blo;
#pragma unroll
    for (int j = 0; j < 8; ++j) {
      ahi[j] = ah0[j]; ahi[j + 8] = ah1[j];
      alo[j] = al0[j]; alo[j + 8] = al1[j];
    }
#pragma unroll
    for (int j = 0; j < 16; ++j) {
      // B 32x16 layout: K = kbase + (j&7) + ((j&8)<<1) + 8*half (strided column of Y)
      const size_t kb = (size_t)(kbase + (j & 7) + ((j & 8) << 1) + (half << 3)) * NSH;
      bhi[j] = yh[kb];
      blo[j] = yl[kb];
    }
    c = wmma3(c, ahi, alo, bhi, blo);
  }

  const float scale = 0.04419417382415922f;  // 1/sqrt(512)
  if (bok) {
#pragma unroll
    for (int r = 0; r < 8; ++r) {
      const int M = rt * 16 + r + (half << 3);  // C layout: VGPR r -> M = r + 8*half
      unsigned short h, l;
      bf16_split(c[r] * scale, h, l);
      phi[(size_t)M * NSH + bcol] = h;
      plo[(size_t)M * NSH + bcol] = l;
    }
  }
}

// ---------------------------------------------------------------------------
// Stage 2 (one launch per l): M=64d rows (g,u), N=512d cols (b,m), K=64.
// Each wave computes a 32x32 macro-tile (2 M-tiles x 2 N-tiles): A and B
// fragments are each reused twice. 8 waves/block, grid (2d, 2d).
// ---------------------------------------------------------------------------
__global__ __launch_bounds__(256) void s2conv_l_kernel(const unsigned short* __restrict__ xhi,
                                                       const unsigned short* __restrict__ xlo,
                                                       const unsigned short* __restrict__ phi,
                                                       const unsigned short* __restrict__ plo,
                                                       float* __restrict__ out,
                                                       int l2, int d, int off) {
  const int lane = threadIdx.x & 31;
  const int wv   = threadIdx.x >> 5;
  const int mt2  = blockIdx.x;               // 0 .. 2d-1 (M tile pair)
  const int nt2  = blockIdx.y * 8 + wv;      // 0 .. 16d-1 (N tile pair)
  const int half = lane >> 4;
  const int lx   = lane & 15;

  const int mrow0 = (mt2 * 2) * 16 + lx;     // row index = g*d + u
  const int mrow1 = mrow0 + 16;
  const int g0 = mrow0 / d, u0 = mrow0 - g0 * d;
  const int g1 = mrow1 / d, u1 = mrow1 - g1 * d;

  const int ncol0 = (nt2 * 2) * 16 + lx;     // col index = b*d + m
  const int ncol1 = ncol0 + 16;
  const int b0 = ncol0 / d, m0 = ncol0 - b0 * d;
  const int b1 = ncol1 / d, m1 = ncol1 - b1 * d;

  // psi planes: [(e*64+g)][i], A stride over e = 64*NSH
  const unsigned short* pA0h = phi + (size_t)g0 * NSH + l2 + u0;
  const unsigned short* pA0l = plo + (size_t)g0 * NSH + l2 + u0;
  const unsigned short* pA1h = phi + (size_t)g1 * NSH + l2 + u1;
  const unsigned short* pA1l = plo + (size_t)g1 * NSH + l2 + u1;
  // x planes: [(b*64+e)][i], B stride over e = NSH
  const unsigned short* pB0h = xhi + (size_t)b0 * (F_IN * NSH) + l2 + m0;
  const unsigned short* pB0l = xlo + (size_t)b0 * (F_IN * NSH) + l2 + m0;
  const unsigned short* pB1h = xhi + (size_t)b1 * (F_IN * NSH) + l2 + m1;
  const unsigned short* pB1l = xlo + (size_t)b1 * (F_IN * NSH) + l2 + m1;

  v8f c00 = {}, c01 = {}, c10 = {}, c11 = {};
#pragma unroll
  for (int kt = 0; kt < 2; ++kt) {           // K = 64 = 2 * 32
    const int kbase = kt * 32;
    v16u a0h, a0l, a1h, a1l, b0h, b0l, b1h, b1l;
#pragma unroll
    for (int j = 0; j < 16; ++j) {
      const size_t ea = (size_t)(kbase + (j & 7) + ((j & 8) << 1) + (half << 3)) * (F_OUT * NSH);
      a0h[j] = pA0h[ea]; a0l[j] = pA0l[ea];
      a1h[j] = pA1h[ea]; a1l[j] = pA1l[ea];
      const size_t eb = (size_t)(kbase + (j & 7) + ((j & 8) << 1) + (half << 3)) * NSH;
      b0h[j] = pB0h[eb]; b0l[j] = pB0l[eb];
      b1h[j] = pB1h[eb]; b1l[j] = pB1l[eb];
    }
    c00 = wmma3(c00, a0h, a0l, b0h, b0l);
    c01 = wmma3(c01, a0h, a0l, b1h, b1l);
    c10 = wmma3(c10, a1h, a1l, b0h, b0l);
    c11 = wmma3(c11, a1h, a1l, b1h, b1l);
  }

  const float alpha = 0.125f;                // 1/sqrt(64)
#pragma unroll
  for (int r = 0; r < 8; ++r) {
    const int M0 = (mt2 * 2) * 16 + r + (half << 3);
    const int M1 = M0 + 16;
    const int ga = M0 / d, ua = M0 - ga * d;
    const int gb = M1 / d, ub = M1 - gb * d;
    out[(size_t)(b0 * F_OUT + ga) * OUTC + off + ua * d + m0] = c00[r] * alpha;
    out[(size_t)(b1 * F_OUT + ga) * OUTC + off + ua * d + m1] = c01[r] * alpha;
    out[(size_t)(b0 * F_OUT + gb) * OUTC + off + ub * d + m0] = c10[r] * alpha;
    out[(size_t)(b1 * F_OUT + gb) * OUTC + off + ub * d + m1] = c11[r] * alpha;
  }
}

// ---------------------------------------------------------------------------
extern "C" void kernel_launch(void* const* d_in, const int* in_sizes, int n_in,
                              void* d_out, int out_size, void* d_ws, size_t ws_size,
                              hipStream_t stream) {
  const float* x = (const float*)d_in[0];   // (512, 64, 121)
  const float* w = (const float*)d_in[1];   // (64, 64, 512)
  const float* Y = (const float*)d_in[2];   // (512, 121)
  float* out = (float*)d_out;               // (512, 64, 1771) fp32

  // Workspace layout (bf16 hi/lo planes), ~25 MB total.
  char* ws = (char*)d_ws;
  unsigned short* xhi = (unsigned short*)(ws);                       // XN
  unsigned short* xlo = xhi + XN;
  unsigned short* whi = xlo + XN;                                    // WN
  unsigned short* wlo = whi + WN;
  unsigned short* yhi = wlo + WN;                                    // YN
  unsigned short* ylo = yhi + YN;
  unsigned short* phi = ylo + YN;                                    // PSIN
  unsigned short* plo = phi + PSIN;

  // Pass 0: one-time fp32 -> bf16 hi/lo splits.
  split_kernel<<<(XN + 255) / 256, 256, 0, stream>>>(x, xhi, xlo, XN);
  split_kernel<<<(WN + 255) / 256, 256, 0, stream>>>(w, whi, wlo, WN);
  split_kernel<<<(YN + 255) / 256, 256, 0, stream>>>(Y, yhi, ylo, YN);

  // Stage 1: psi GEMM (bf16-plane in, bf16-plane out).
  psi_gemm_kernel<<<dim3(64, 8), dim3(128), 0, stream>>>(whi, wlo, yhi, ylo, phi, plo);

  // Stage 2: per-l block GEMMs.
  int off = 0;
  for (int l = 0; l <= 10; ++l) {
    const int d = 2 * l + 1;
    s2conv_l_kernel<<<dim3(2 * d, 2 * d), dim3(256), 0, stream>>>(xhi, xlo, phi, plo,
                                                                  out, l * l, d, off);
    off += d * d;
  }
  (void)in_sizes; (void)n_in; (void)out_size; (void)ws_size;
}
